// ResNet18SingleBranchLSTM_66305705115717
// MI455X (gfx1250) — matrix operand order repaired and checked by reading, not compiled, not run on hardware
//
#include <hip/hip_runtime.h>
#include <hip/hip_bf16.h>

// ---------------------------------------------------------------------------
// CDNA5 (gfx1250) implementation: CNN -> BiLSTM -> MLP -> softmax
// All GEMM-shaped work uses v_wmma_f32_16x16x32_f16 (wave32 WMMA).
// Conv kernels templated on (Cin, Cout, KW); all LDS staging vectorized
// (float4 global loads -> f16 -> b64/b16 LDS stores).
// ---------------------------------------------------------------------------

typedef __attribute__((ext_vector_type(16))) _Float16 v16h;
typedef __attribute__((ext_vector_type(8)))  float    v8f;

struct alignas(16) h8 { _Float16 d[8]; };
struct alignas(8)  h4 { _Float16 d[4]; };
union U16 { v16h v; h8 p[2]; };

#define WMMA_F16(A, B, C) \
    __builtin_amdgcn_wmma_f32_16x16x32_f16(false, (A), false, (B), (short)0, (C), false, false)

static __device__ __forceinline__ float sigmoidf_(float x) {
    return 1.0f / (1.0f + __expf(-x));
}

static __device__ __forceinline__ h4 cvt4(float4 f) {
    h4 h;
    h.d[0] = (_Float16)f.x; h.d[1] = (_Float16)f.y;
    h.d[2] = (_Float16)f.z; h.d[3] = (_Float16)f.w;
    return h;
}

// Model constants
#define BATCH   256
#define AXES    6
#define TIN     512
#define TCONV   254      // after VALID conv1 (508) + maxpool/2
#define FEAT    384      // 6 * 64
#define UNITS   128
#define GDIM    512      // 4*UNITS
#define NCLS    100
#define EPS_BN  1e-3f

// ---------------------------------------------------------------------------
// conv1 (1x5 VALID, Cin=1 -> 32) + BN + ReLU + maxpool(2) fused, scalar f32.
// x: [B][6][512][1], out: [B][6][254][32]  (NHWC)
// ---------------------------------------------------------------------------
__global__ void __launch_bounds__(256)
k_conv1_pool(const float* __restrict__ x, const float* __restrict__ w,
             const float* __restrict__ cb, const float* __restrict__ g,
             const float* __restrict__ bb, const float* __restrict__ m,
             const float* __restrict__ vv, float* __restrict__ out)
{
    size_t i = (size_t)blockIdx.x * 256 + threadIdx.x;
    const size_t ntot = (size_t)BATCH * AXES * TCONV * 32;
    if (i >= ntot) return;
    int co = (int)(i & 31);
    size_t rest = i >> 5;
    int wo = (int)(rest % TCONV); rest /= TCONV;
    int h  = (int)(rest % AXES);
    int b  = (int)(rest / AXES);
    const float* xr = x + ((size_t)(b * AXES + h)) * TIN;
    float s    = g[co] * rsqrtf(vv[co] + EPS_BN);
    float beta = (cb[co] - m[co]) * s + bb[co];
    float best = 0.0f;  // relu floor
    #pragma unroll
    for (int p = 0; p < 2; ++p) {
        int wp = 2 * wo + p;             // conv output col in [0,508)
        float acc = 0.f;
        #pragma unroll
        for (int k = 0; k < 5; ++k) acc += xr[wp + k] * w[k * 32 + co];
        float y = fmaxf(acc * s + beta, 0.0f);
        best = fmaxf(best, y);
    }
    out[((size_t)(b * AXES + h) * TCONV + wo) * 32 + co] = best;
}

// ---------------------------------------------------------------------------
// Implicit-GEMM conv (1xKW along W, SAME for KW=3, 1x1 for KW=1) + BN
// (+ optional residual) (+ optional ReLU).  WMMA f16 -> f32.
// in : [B][6][254][Cin], out/resid: [B][6][254][Cout]
// w  : [KW][Cin][Cout] row-major (HWIO), cb: conv bias
// Block: 128 threads (4 waves), owns 64 consecutive w positions.
// ---------------------------------------------------------------------------
template <int CIN, int COUT, int KW>
__global__ void __launch_bounds__(128)
k_conv_bn(const float* __restrict__ in, const float* __restrict__ w,
          const float* __restrict__ cb, const float* __restrict__ bng,
          const float* __restrict__ bnb, const float* __restrict__ bnm,
          const float* __restrict__ bnv, const float* __restrict__ resid,
          float* __restrict__ out, int relu)
{
    constexpr int PAD = (KW == 3) ? 1 : 0;
    constexpr int K   = KW * CIN;
    constexpr int KCE = K / 32;               // K chunks of 32
    constexpr int NTN = COUT / 16;            // N tiles
    constexpr int CSH = (CIN == 64) ? 6 : 5;  // log2(CIN)
    constexpr int CMSK = CIN - 1;

    __shared__ alignas(16) _Float16 strip[(64 + 2 * PAD) * 64];  // [rows][64], stride 64
    __shared__ alignas(16) _Float16 wT[COUT * K];                // [Cout][K], stride K

    const int tid = threadIdx.x;
    const int wb  = blockIdx.x & 3;
    const int bh  = blockIdx.x >> 2;
    const int h   = bh % AXES, b = bh / AXES;
    const int w0  = wb * 64;
    const size_t rowbase = (size_t)(b * AXES + h) * TCONV;

    // Stage activation strip (float4 loads -> f16 -> b64 LDS stores)
    constexpr int NQ   = CIN / 4;             // float4 chunks per row
    constexpr int NQSH = (CIN == 64) ? 4 : 3; // log2(NQ)
    for (int i = tid; i < (64 + 2 * PAD) * NQ; i += 128) {
        int r  = i >> NQSH;
        int ci = (i & (NQ - 1)) * 4;
        int wg = w0 + r - PAD;
        h4 hv; hv.d[0] = hv.d[1] = hv.d[2] = hv.d[3] = (_Float16)0.f;
        if (wg >= 0 && wg < TCONV)
            hv = cvt4(*reinterpret_cast<const float4*>(&in[(rowbase + wg) * CIN + ci]));
        *reinterpret_cast<h4*>(&strip[r * 64 + ci]) = hv;
    }
    // Stage transposed weights: wT[co][k] = w[k][co]  (float4 along co)
    constexpr int CQ = COUT / 4;
    for (int i = tid; i < K * CQ; i += 128) {
        int k  = i / CQ;                      // CQ is a power of two
        int cq = (i - k * CQ) * 4;
        h4 hv = cvt4(*reinterpret_cast<const float4*>(&w[k * COUT + cq]));
        wT[(cq + 0) * K + k] = hv.d[0];
        wT[(cq + 1) * K + k] = hv.d[1];
        wT[(cq + 2) * K + k] = hv.d[2];
        wT[(cq + 3) * K + k] = hv.d[3];
    }
    __syncthreads();

    const int lane = tid & 31, wid = tid >> 5;
    const int half = lane >> 4, lr = lane & 15;

    for (int t = wid; t < 4 * NTN; t += 4) {
        const int mi = t & 3, ni = t >> 2;
        v8f acc = {0, 0, 0, 0, 0, 0, 0, 0};
        const int mloc = mi * 16 + lr;
        const int n = ni * 16 + lr;
        #pragma unroll
        for (int kc = 0; kc < KCE; ++kc) {
            U16 ua, ub;
            {   // A fragment: two 8-element runs per the wave32 16x32 layout
                int k0 = kc * 32 + half * 8;
                int kw0 = k0 >> CSH, ci0 = k0 & CMSK;
                ua.p[0] = *reinterpret_cast<const h8*>(&strip[(mloc + kw0) * 64 + ci0]);
                int k1 = kc * 32 + 16 + half * 8;
                int kw1 = k1 >> CSH, ci1 = k1 & CMSK;
                ua.p[1] = *reinterpret_cast<const h8*>(&strip[(mloc + kw1) * 64 + ci1]);
            }
            {   // B fragment: two 8-element runs matching the A layout
                int kb = kc * 32 + half * 8;
                ub.p[0] = *reinterpret_cast<const h8*>(&wT[n * K + kb]);
                ub.p[1] = *reinterpret_cast<const h8*>(&wT[n * K + kb + 16]);
            }
            acc = WMMA_F16(ua.v, ub.v, acc);
        }
        // Epilogue: fold conv bias into BN, residual, relu
        const int co = n;
        const float s    = bng[co] * rsqrtf(bnv[co] + EPS_BN);
        const float beta = (cb[co] - bnm[co]) * s + bnb[co];
        #pragma unroll
        for (int e = 0; e < 8; ++e) {
            int wg = w0 + mi * 16 + e + 8 * half;
            if (wg < TCONV) {
                float val = acc[e] * s + beta;
                if (resid) val += resid[(rowbase + wg) * COUT + co];
                if (relu)  val = fmaxf(val, 0.f);
                out[(rowbase + wg) * COUT + co] = val;
            }
        }
    }
}

// ---------------------------------------------------------------------------
// seq[(b*254+t)][h*64+c] = conv_out[b][h][t][c]   (float4 both sides)
// ---------------------------------------------------------------------------
__global__ void __launch_bounds__(256)
k_transpose_seq(const float* __restrict__ in, float* __restrict__ seq)
{
    size_t q = (size_t)blockIdx.x * 256 + threadIdx.x;   // float4 index
    const size_t nq = (size_t)BATCH * TCONV * FEAT / 4;
    if (q >= nq) return;
    size_t i = q * 4;
    int f = (int)(i % FEAT);
    size_t bt = i / FEAT;
    int t = (int)(bt % TCONV);
    int b = (int)(bt / TCONV);
    int h = f >> 6, c = f & 63;                           // 4-runs never cross h
    *reinterpret_cast<float4*>(&seq[i]) =
        *reinterpret_cast<const float4*>(&in[((size_t)(b * AXES + h) * TCONV + t) * 64 + c]);
}

// ---------------------------------------------------------------------------
// Generic tiled WMMA GEMM: C[M,N] = A[M,K] * B[K,N] (+bias) (+relu)
// Block 256 threads (8 waves), 64x64 tile, K staged in 32-chunks (f16 LDS).
// Staging is float4-vectorized (all call sites have K%32==0, N%4==0).
// ---------------------------------------------------------------------------
__global__ void __launch_bounds__(256)
k_gemm(const float* __restrict__ A, const float* __restrict__ B,
       const float* __restrict__ bias, float* __restrict__ C,
       int M, int N, int K, int relu)
{
    __shared__ alignas(16) _Float16 As[64 * 32];   // [m][k]
    __shared__ alignas(16) _Float16 Bt[64 * 32];   // [n][k] (transposed)

    const int tid = threadIdx.x;
    const int n0 = blockIdx.x * 64, m0 = blockIdx.y * 64;
    const int lane = tid & 31, wid = tid >> 5;
    const int half = lane >> 4, lr = lane & 15;
    const int mi  = wid & 3;
    const int ni0 = wid >> 2, ni1 = ni0 + 2;   // this wave's two N-tiles

    v8f acc0 = {0, 0, 0, 0, 0, 0, 0, 0};
    v8f acc1 = {0, 0, 0, 0, 0, 0, 0, 0};
    const int kcEnd = (K + 31) >> 5;

    for (int kc = 0; kc < kcEnd; ++kc) {
        __syncthreads();
        // A: 64 rows x 32 k  = 512 float4 chunks
        for (int i = tid; i < 512; i += 256) {
            int r = i >> 3, kq = (i & 7) * 4;
            int gm = m0 + r, gk = kc * 32 + kq;
            h4 hv;
            if (gm < M && gk + 4 <= K) {
                hv = cvt4(*reinterpret_cast<const float4*>(&A[(size_t)gm * K + gk]));
            } else {
                #pragma unroll
                for (int j = 0; j < 4; ++j)
                    hv.d[j] = (gm < M && gk + j < K) ? (_Float16)A[(size_t)gm * K + gk + j]
                                                     : (_Float16)0.f;
            }
            *reinterpret_cast<h4*>(&As[r * 32 + kq]) = hv;
        }
        // B: transpose-stage, float4 along n
        for (int i = tid; i < 512; i += 256) {
            int k = i & 31, nq = (i >> 5) * 4;
            int gn = n0 + nq, gk = kc * 32 + k;
            h4 hv;
            if (gk < K && gn + 4 <= N) {
                hv = cvt4(*reinterpret_cast<const float4*>(&B[(size_t)gk * N + gn]));
            } else {
                #pragma unroll
                for (int j = 0; j < 4; ++j)
                    hv.d[j] = (gk < K && gn + j < N) ? (_Float16)B[(size_t)gk * N + gn + j]
                                                     : (_Float16)0.f;
            }
            Bt[(nq + 0) * 32 + k] = hv.d[0];
            Bt[(nq + 1) * 32 + k] = hv.d[1];
            Bt[(nq + 2) * 32 + k] = hv.d[2];
            Bt[(nq + 3) * 32 + k] = hv.d[3];
        }
        __syncthreads();
        // Prefetch next K-chunk into cache (global_prefetch_b8)
        if (kc + 1 < kcEnd) {
            if (tid < 64) {
                int gm = m0 + tid;
                if (gm < M) __builtin_prefetch(&A[(size_t)gm * K + (kc + 1) * 32], 0, 1);
            } else if (tid < 96) {
                int gk = (kc + 1) * 32 + (tid & 31);
                if (gk < K) __builtin_prefetch(&B[(size_t)gk * N + n0], 0, 1);
            }
        }
        U16 ua, ub0, ub1;
        ua.p[0]  = *reinterpret_cast<const h8*>(&As[(mi * 16 + lr) * 32 + half * 8]);
        ua.p[1]  = *reinterpret_cast<const h8*>(&As[(mi * 16 + lr) * 32 + 16 + half * 8]);
        ub0.p[0] = *reinterpret_cast<const h8*>(&Bt[(ni0 * 16 + lr) * 32 + half * 8]);
        ub0.p[1] = *reinterpret_cast<const h8*>(&Bt[(ni0 * 16 + lr) * 32 + 16 + half * 8]);
        ub1.p[0] = *reinterpret_cast<const h8*>(&Bt[(ni1 * 16 + lr) * 32 + half * 8]);
        ub1.p[1] = *reinterpret_cast<const h8*>(&Bt[(ni1 * 16 + lr) * 32 + 16 + half * 8]);
        acc0 = WMMA_F16(ua.v, ub0.v, acc0);
        acc1 = WMMA_F16(ua.v, ub1.v, acc1);
    }

    #pragma unroll
    for (int e = 0; e < 8; ++e) {
        int r = m0 + mi * 16 + e + 8 * half;
        if (r < M) {
            int c0 = n0 + ni0 * 16 + lr;
            if (c0 < N) {
                float v = acc0[e] + (bias ? bias[c0] : 0.f);
                if (relu) v = fmaxf(v, 0.f);
                C[(size_t)r * N + c0] = v;
            }
            int c1 = n0 + ni1 * 16 + lr;
            if (c1 < N) {
                float v = acc1[e] + (bias ? bias[c1] : 0.f);
                if (relu) v = fmaxf(v, 0.f);
                C[(size_t)r * N + c1] = v;
            }
        }
    }
}

// ---------------------------------------------------------------------------
// whT[n][k] = (f16) Wh[k][n]   (Wh: [128][512] -> whT: [512][128])
// ---------------------------------------------------------------------------
__global__ void __launch_bounds__(256)
k_prep_whT(const float* __restrict__ Wh, _Float16* __restrict__ whT)
{
    int i = blockIdx.x * 256 + threadIdx.x;   // 512*128 = 65536
    if (i >= GDIM * UNITS) return;
    int n = i >> 7, k = i & 127;
    whT[i] = (_Float16)Wh[(size_t)k * GDIM + n];
}

// ---------------------------------------------------------------------------
// Recurrent LSTM: one 1024-thread workgroup per direction.
// H kept in LDS as f16 (exact WMMA A operand).  C and Z in global scratch.
// Per step: Z = xz[:,t,:] + H @ Wh  (16x32 WMMA tiles, 32 waves), then gates.
// ---------------------------------------------------------------------------
__global__ void __launch_bounds__(1024)
k_lstm(const float* __restrict__ xzf, const float* __restrict__ xzb,
       const _Float16* __restrict__ whT2, float* __restrict__ Cst2,
       float* __restrict__ Zsc2, float* __restrict__ hcat)
{
    __shared__ alignas(16) _Float16 Hs[BATCH * UNITS];   // 64 KB

    const int dir = blockIdx.x;
    const float*     xz = dir ? xzb : xzf;
    const _Float16*  W  = whT2 + (size_t)dir * GDIM * UNITS;
    float* Cs = Cst2 + (size_t)dir * BATCH * UNITS;
    float* Z  = Zsc2 + (size_t)dir * BATCH * GDIM;

    const int tid = threadIdx.x;
    for (int i = tid; i < BATCH * UNITS; i += 1024) { Hs[i] = (_Float16)0.f; Cs[i] = 0.f; }
    __syncthreads();

    const int lane = tid & 31, wid = tid >> 5;
    const int half = lane >> 4, lr = lane & 15;

    for (int s = 0; s < TCONV; ++s) {
        const int t = dir ? (TCONV - 1 - s) : s;

        // Phase 1: Z = xz_t + H @ Wh   (512 tiles over 32 waves)
        for (int tt = wid; tt < 512; tt += 32) {
            const int mi = tt & 15, ni = tt >> 4;
            const int gcol = ni * 16 + lr;
            v8f acc;
            #pragma unroll
            for (int e = 0; e < 8; ++e) {
                int b = mi * 16 + e + 8 * half;
                acc[e] = xz[((size_t)b * TCONV + t) * GDIM + gcol];
            }
            #pragma unroll
            for (int kc = 0; kc < 4; ++kc) {
                U16 ua, ub;
                ua.p[0] = *reinterpret_cast<const h8*>(&Hs[(mi * 16 + lr) * UNITS + kc * 32 + half * 8]);
                ua.p[1] = *reinterpret_cast<const h8*>(&Hs[(mi * 16 + lr) * UNITS + kc * 32 + 16 + half * 8]);
                ub.p[0] = *reinterpret_cast<const h8*>(&W[(size_t)gcol * UNITS + kc * 32 + half * 8]);
                ub.p[1] = *reinterpret_cast<const h8*>(&W[(size_t)gcol * UNITS + kc * 32 + 16 + half * 8]);
                acc = WMMA_F16(ua.v, ub.v, acc);
            }
            #pragma unroll
            for (int e = 0; e < 8; ++e) {
                int b = mi * 16 + e + 8 * half;
                Z[(size_t)b * GDIM + gcol] = acc[e];
            }
        }
        __syncthreads();   // all H reads + Z writes done

        // Phase 2: gates, C update, next H (f16 to LDS), h to global
        for (int i = tid; i < BATCH * UNITS; i += 1024) {
            int b = i >> 7, u = i & 127;
            const float* zr = Z + (size_t)b * GDIM;
            float zi = zr[u], zf = zr[UNITS + u], zg = zr[2 * UNITS + u], zo = zr[3 * UNITS + u];
            float c  = Cs[i];
            float cn = sigmoidf_(zf) * c + sigmoidf_(zi) * tanhf(zg);
            float hh = sigmoidf_(zo) * tanhf(cn);
            Cs[i] = cn;
            Hs[i] = (_Float16)hh;
            hcat[((size_t)b * TCONV + t) * (2 * UNITS) + dir * UNITS + u] = hh;
        }
        __syncthreads();   // H stable before next step's reads
    }
}

// ---------------------------------------------------------------------------
// pooled[b][j] = mean_t hcat[b][t][j]
// ---------------------------------------------------------------------------
__global__ void __launch_bounds__(256)
k_meanpool(const float* __restrict__ hcat, float* __restrict__ pooled)
{
    int i = blockIdx.x * 256 + threadIdx.x;   // 256*256
    int b = i >> 8, j = i & 255;
    float s = 0.f;
    for (int t = 0; t < TCONV; ++t)
        s += hcat[((size_t)b * TCONV + t) * (2 * UNITS) + j];
    pooled[i] = s * (1.0f / (float)TCONV);
}

// ---------------------------------------------------------------------------
// Row softmax over 100 classes
// ---------------------------------------------------------------------------
__global__ void __launch_bounds__(128)
k_softmax(const float* __restrict__ logits, float* __restrict__ out)
{
    __shared__ float red[128];
    int r = blockIdx.x, j = threadIdx.x;
    float v = (j < NCLS) ? logits[(size_t)r * NCLS + j] : -3.4e38f;
    red[j] = v; __syncthreads();
    for (int s = 64; s > 0; s >>= 1) { if (j < s) red[j] = fmaxf(red[j], red[j + s]); __syncthreads(); }
    float mx = red[0]; __syncthreads();
    float e = (j < NCLS) ? __expf(v - mx) : 0.f;
    red[j] = e; __syncthreads();
    for (int s = 64; s > 0; s >>= 1) { if (j < s) red[j] += red[j + s]; __syncthreads(); }
    float inv = 1.0f / red[0];
    if (j < NCLS) out[(size_t)r * NCLS + j] = e * inv;
}

// ---------------------------------------------------------------------------
// Host launcher
// ---------------------------------------------------------------------------
extern "C" void kernel_launch(void* const* d_in, const int* in_sizes, int n_in,
                              void* d_out, int out_size, void* d_ws, size_t ws_size,
                              hipStream_t stream) {
    (void)in_sizes; (void)n_in; (void)out_size; (void)ws_size;
    #define FP(i) ((const float*)d_in[i])
    // Input index map (setup_inputs flattening order):
    // 0:x | 1,2:conv1 w,b | 3..6:bn1 g,b,m,v
    // b1a: 7,8 c1 | 9..12 bn1 | 13,14 c2 | 15..18 bn2
    // b1b: 19..30 | b2a: 31..42 + 43,44 down + 45..48 dbn | b2b: 49..60
    // 61..63 lstm_f Wx,Wh,b | 64..66 lstm_b | 67,68 dense1 | 69,70 user

    const size_t SZ32  = (size_t)BATCH * AXES * TCONV * 32;   // 12,484,608
    const size_t SZ64  = SZ32 * 2;
    const size_t XZSZ  = (size_t)BATCH * TCONV * GDIM;

    float* W = (float*)d_ws;
    float* BUF_A = W;
    float* BUF_B = BUF_A + SZ32;
    float* BUF_C = BUF_B + SZ32;
    float* BUF_D = BUF_C + SZ32;
    float* BUF_E = BUF_D + SZ64;
    float* BUF_F = BUF_E + SZ64;
    float* XZ_B  = BUF_F + SZ64;
    _Float16* WHT = (_Float16*)(XZ_B + XZSZ);             // 2 * 512*128 halfs
    float* ZSC    = (float*)(WHT + 2 * GDIM * UNITS);
    float* CST    = ZSC + 2 * BATCH * GDIM;
    float* POOLED = CST + 2 * BATCH * UNITS;
    float* D1     = POOLED + BATCH * 2 * UNITS;
    float* LOGITS = D1 + BATCH * 128;
    // Lifetime-based aliases (earlier buffers dead by the time these are used)
    float* SEQ  = BUF_A;   // <= A+B+C region
    float* XZ_F = BUF_E;   // <= E+F region
    float* HCAT = BUF_D;   // <= D region
    float* OUT  = (float*)d_out;

    const dim3 convGrid(BATCH * AXES * 4);

    // 1) conv1 + bn + relu + maxpool  -> BUF_A (32ch)
    k_conv1_pool<<<(unsigned)(SZ32 / 256), 256, 0, stream>>>(
        FP(0), FP(1), FP(2), FP(3), FP(4), FP(5), FP(6), BUF_A);

    // 2) b1a
    k_conv_bn<32, 32, 3><<<convGrid, 128, 0, stream>>>(BUF_A, FP(7),  FP(8),  FP(9),  FP(10), FP(11), FP(12), nullptr, BUF_B, 1);
    k_conv_bn<32, 32, 3><<<convGrid, 128, 0, stream>>>(BUF_B, FP(13), FP(14), FP(15), FP(16), FP(17), FP(18), BUF_A,   BUF_C, 1);
    // 3) b1b
    k_conv_bn<32, 32, 3><<<convGrid, 128, 0, stream>>>(BUF_C, FP(19), FP(20), FP(21), FP(22), FP(23), FP(24), nullptr, BUF_B, 1);
    k_conv_bn<32, 32, 3><<<convGrid, 128, 0, stream>>>(BUF_B, FP(25), FP(26), FP(27), FP(28), FP(29), FP(30), BUF_C,   BUF_A, 1);
    // 4) b2a (downsample residual, then convs)
    k_conv_bn<32, 64, 1><<<convGrid, 128, 0, stream>>>(BUF_A, FP(43), FP(44), FP(45), FP(46), FP(47), FP(48), nullptr, BUF_D, 0);
    k_conv_bn<32, 64, 3><<<convGrid, 128, 0, stream>>>(BUF_A, FP(31), FP(32), FP(33), FP(34), FP(35), FP(36), nullptr, BUF_E, 1);
    k_conv_bn<64, 64, 3><<<convGrid, 128, 0, stream>>>(BUF_E, FP(37), FP(38), FP(39), FP(40), FP(41), FP(42), BUF_D,   BUF_F, 1);
    // 5) b2b
    k_conv_bn<64, 64, 3><<<convGrid, 128, 0, stream>>>(BUF_F, FP(49), FP(50), FP(51), FP(52), FP(53), FP(54), nullptr, BUF_E, 1);
    k_conv_bn<64, 64, 3><<<convGrid, 128, 0, stream>>>(BUF_E, FP(55), FP(56), FP(57), FP(58), FP(59), FP(60), BUF_F,   BUF_D, 1);

    // 6) transpose to sequence layout (b,t,f)
    k_transpose_seq<<<(unsigned)(((size_t)BATCH * TCONV * FEAT / 4) / 256), 256, 0, stream>>>(BUF_D, SEQ);

    // 7) xz = seq @ Wx + b  (both directions)
    {
        const int M = BATCH * TCONV;
        dim3 g(GDIM / 64, M / 64);
        k_gemm<<<g, 256, 0, stream>>>(SEQ, FP(61), FP(63), XZ_F, M, GDIM, FEAT, 0);
        k_gemm<<<g, 256, 0, stream>>>(SEQ, FP(64), FP(66), XZ_B, M, GDIM, FEAT, 0);
    }

    // 8) pre-transpose Wh to f16 [512][128] for direct B-fragment loads
    k_prep_whT<<<GDIM * UNITS / 256, 256, 0, stream>>>(FP(62), WHT);
    k_prep_whT<<<GDIM * UNITS / 256, 256, 0, stream>>>(FP(65), WHT + GDIM * UNITS);

    // 9) recurrent bidirectional LSTM -> HCAT
    k_lstm<<<2, 1024, 0, stream>>>(XZ_F, XZ_B, WHT, CST, ZSC, HCAT);

    // 10) mean pool over time
    k_meanpool<<<(BATCH * 2 * UNITS) / 256, 256, 0, stream>>>(HCAT, POOLED);

    // 11) dense1 + relu
    k_gemm<<<dim3(2, 4), 256, 0, stream>>>(POOLED, FP(67), FP(68), D1, BATCH, 128, 2 * UNITS, 1);
    // 12) classifier
    k_gemm<<<dim3(2, 4), 256, 0, stream>>>(D1, FP(69), FP(70), LOGITS, BATCH, NCLS, 128, 0);
    // 13) softmax -> d_out
    k_softmax<<<BATCH, 128, 0, stream>>>(LOGITS, OUT);
    #undef FP
}
